// CombinedModel_1872605741272
// MI455X (gfx1250) — matrix operand order repaired and checked by reading, not compiled, not run on hardware
//
#include <hip/hip_runtime.h>

#define N_ATOMS 50000
#define E_INNER 150000
#define N_AA    8000
#define E_AA    24000
#define IN_C    42
#define HD      600
#define EDGE_D  5
#define EMB     32
#define MLP_H   128
#define NLAYERS 4

typedef __attribute__((ext_vector_type(16))) _Float16 v16h;
typedef __attribute__((ext_vector_type(8)))  _Float16 v8h;
typedef __attribute__((ext_vector_type(8)))  float    v8f;

// ---------------------------------------------------------------- utilities
__device__ __forceinline__ float warp_sum32(float v) {
#pragma unroll
  for (int m = 16; m >= 1; m >>= 1) v += __shfl_xor(v, m, 32);
  return v;
}

__device__ __forceinline__ void atomicMaxF(float* addr, float val) {
  unsigned* ua = (unsigned*)addr;
  unsigned old = *ua;
  while (__uint_as_float(old) < val) {
    unsigned assumed = old;
    old = atomicCAS(ua, assumed, __float_as_uint(val));
    if (old == assumed) break;
  }
}

__global__ void fill_f32(float* p, float v, long n) {
  long i = (long)blockIdx.x * blockDim.x + threadIdx.x;
  long st = (long)gridDim.x * blockDim.x;
  for (; i < n; i += st) p[i] = v;
}

// ---------------------------------------------------------------- WMMA GEMM
// C[M,N] = act( A[M,K] * B + bias ), A row-major f32 (converted to f16 in LDS).
// bTrans==0: B is [K,N] row-major (ldb=N).  bTrans==1: B is [N,K] row-major
// (ldb=K), i.e. computes A @ B^T  (PyTorch x @ W.T with W=[N,K]).
// 256x64x32 block tile, 8 waves, each wave owns 32 rows x 64 cols:
//   2 A fragments x 4 B fragments -> 8 WMMA / K-step, B fragments reused 2x.
// Double-buffered LDS software pipeline: next tile's global loads issue
// before the current tile's WMMA chain; LDS stores land after the WMMAs.
#define BM 256
#define BN 64
#define BK 32
#define APITCH 40
#define BPITCH 40
#define ATILE (BM * APITCH)
#define BTILE (BN * BPITCH)

__global__ void __launch_bounds__(256)
gemm_wmma_f16(const float* __restrict__ A, const float* __restrict__ B,
              const float* __restrict__ bias, float* __restrict__ C,
              int M, int N, int K, int ldb, int bTrans, int relu) {
  __shared__ _Float16 As[2 * ATILE];
  __shared__ _Float16 Bs[2 * BTILE];  // stored transposed: Bs[n*BPITCH + k]

  const int tid  = threadIdx.x;
  const int wave = tid >> 5;
  const int lane = tid & 31;
  const int hf   = lane >> 4;   // which 16-lane half
  const int l16  = lane & 15;

  const int m0 = blockIdx.y * BM;
  const int n0 = blockIdx.x * BN;

  const bool aRowsFull = (m0 + BM) <= M;   // all 256 rows valid
  const bool bColsFull = (n0 + BN) <= N;   // all 64 cols valid

  // per-thread staging coordinates
  const int ar_r = tid;                // A: one full 32-col row per thread
  const int bt_n = tid >> 2;           // B fast path (bTrans): row n, 8 k's
  const int bt_k = (tid & 3) * 8;
  const int bn_k = tid >> 3;           // B fast path (!bTrans): row k, 8 n's
  const int bn_n = (tid & 7) * 8;

  // ---- combined load+store staging (prologue / edge tiles) ----
  auto stageA = [&](int k0, _Float16* buf, bool fast) {
    if (fast) {
      const float* ga = &A[(long)(m0 + ar_r) * K + k0];
      _Float16 tmp[32];
#pragma unroll
      for (int i = 0; i < 16; ++i) {
        float2 v = *(const float2*)(ga + 2 * i);
        tmp[2 * i]     = (_Float16)v.x;
        tmp[2 * i + 1] = (_Float16)v.y;
      }
#pragma unroll
      for (int q = 0; q < 4; ++q) {
        v8h pk;
#pragma unroll
        for (int i = 0; i < 8; ++i) pk[i] = tmp[q * 8 + i];
        *(v8h*)&buf[ar_r * APITCH + q * 8] = pk;
      }
    } else {
      int gr = m0 + ar_r;
#pragma unroll 1
      for (int c = 0; c < 32; ++c) {
        int gc = k0 + c;
        float v = 0.0f;
        if (gr < M && gc < K) v = A[(long)gr * K + gc];
        buf[ar_r * APITCH + c] = (_Float16)v;
      }
    }
  };
  auto stageB = [&](int k0, _Float16* buf, bool fast) {
    if (fast) {
      if (bTrans) {
        const float* gb = &B[(long)(n0 + bt_n) * ldb + k0 + bt_k];
        v8h pk;
#pragma unroll
        for (int i = 0; i < 4; ++i) {
          float2 v = *(const float2*)(gb + 2 * i);
          pk[2 * i]     = (_Float16)v.x;
          pk[2 * i + 1] = (_Float16)v.y;
        }
        *(v8h*)&buf[bt_n * BPITCH + bt_k] = pk;
      } else {
        const float* gb = &B[(long)(k0 + bn_k) * ldb + n0 + bn_n];
#pragma unroll
        for (int i = 0; i < 4; ++i) {
          float2 v = *(const float2*)(gb + 2 * i);
          buf[(bn_n + 2 * i) * BPITCH + bn_k]     = (_Float16)v.x;
          buf[(bn_n + 2 * i + 1) * BPITCH + bn_k] = (_Float16)v.y;
        }
      }
    } else {
#pragma unroll 1
      for (int i = 0; i < 8; ++i) {
        int e = tid * 8 + i;
        int n = e >> 5, k = e & 31;
        int gn = n0 + n, gk = k0 + k;
        float v = 0.0f;
        if (gn < N && gk < K)
          v = bTrans ? B[(long)gn * ldb + gk] : B[(long)gk * ldb + gn];
        buf[n * BPITCH + k] = (_Float16)v;
      }
    }
  };

  v8f acc0[4] = {};  // rows wave*32 + 0..15
  v8f acc1[4] = {};  // rows wave*32 + 16..31

  // prologue: stage k0 = 0 into buffer 0
  {
    bool kFull0 = BK <= K;
    stageA(0, As, aRowsFull && kFull0);
    stageB(0, Bs, bColsFull && kFull0);
  }

  int cur = 0;
  for (int k0 = 0; k0 < K; k0 += BK, cur ^= 1) {
    const int kn = k0 + BK;
    const bool haveNext = kn < K;
    bool fa = false, fb = false;
    float ar[32];
    float br[8];

    __syncthreads();

    // ---- issue next tile's global loads into registers (fast paths) ----
    if (haveNext) {
      const bool knFull = (kn + BK) <= K;
      fa = aRowsFull && knFull;
      fb = bColsFull && knFull;
      if (fa) {
        const float* ga = &A[(long)(m0 + ar_r) * K + kn];
#pragma unroll
        for (int i = 0; i < 16; ++i) {
          float2 v = *(const float2*)(ga + 2 * i);
          ar[2 * i] = v.x; ar[2 * i + 1] = v.y;
        }
      }
      if (fb) {
        const float* gb = bTrans ? &B[(long)(n0 + bt_n) * ldb + kn + bt_k]
                                 : &B[(long)(kn + bn_k) * ldb + n0 + bn_n];
#pragma unroll
        for (int i = 0; i < 4; ++i) {
          float2 v = *(const float2*)(gb + 2 * i);
          br[2 * i] = v.x; br[2 * i + 1] = v.y;
        }
      }
    }

    // ---- compute from LDS buffer `cur` ----
    const _Float16* Asb = As + cur * ATILE;
    const _Float16* Bsb = Bs + cur * BTILE;

    // A fragments (16x32 f16): lanes 0-15 M=lane; half 0 -> K 0..7 / 16..23,
    // half 1 -> K 8..15 / 24..31  (ISA 7.12.2 16-bit A layout)
    v16h a0, a1;
    {
      const _Float16* ap0 = &Asb[(wave * 32 + l16) * APITCH];
      const _Float16* ap1 = &Asb[(wave * 32 + 16 + l16) * APITCH];
#pragma unroll
      for (int i = 0; i < 8; ++i) {
        a0[i]     = ap0[hf * 8 + i];
        a0[i + 8] = ap0[16 + hf * 8 + i];
        a1[i]     = ap1[hf * 8 + i];
        a1[i + 8] = ap1[16 + hf * 8 + i];
      }
    }
#pragma unroll
    for (int t = 0; t < 4; ++t) {
      // B fragment (32x16): lane col = l16, K = hf*8 + e / 16 + hf*8 + e (same as A)
      v16h b;
      const _Float16* bp = &Bsb[(t * 16 + l16) * BPITCH + hf * 8];
#pragma unroll
      for (int i = 0; i < 8; ++i) { b[i] = bp[i]; b[i + 8] = bp[16 + i]; }
      acc0[t] = __builtin_amdgcn_wmma_f32_16x16x32_f16(
          false, a0, false, b, (short)0, acc0[t], false, false);
      acc1[t] = __builtin_amdgcn_wmma_f32_16x16x32_f16(
          false, a1, false, b, (short)0, acc1[t], false, false);
    }

    // ---- store next tile into the other LDS buffer ----
    if (haveNext) {
      _Float16* Asn = As + (cur ^ 1) * ATILE;
      _Float16* Bsn = Bs + (cur ^ 1) * BTILE;
      if (fa) {
#pragma unroll
        for (int q = 0; q < 4; ++q) {
          v8h pk;
#pragma unroll
          for (int i = 0; i < 8; ++i) pk[i] = (_Float16)ar[q * 8 + i];
          *(v8h*)&Asn[ar_r * APITCH + q * 8] = pk;
        }
      } else {
        stageA(kn, Asn, false);
      }
      if (fb) {
        if (bTrans) {
          v8h pk;
#pragma unroll
          for (int i = 0; i < 8; ++i) pk[i] = (_Float16)br[i];
          *(v8h*)&Bsn[bt_n * BPITCH + bt_k] = pk;
        } else {
#pragma unroll
          for (int i = 0; i < 8; ++i)
            Bsn[(bn_n + i) * BPITCH + bn_k] = (_Float16)br[i];
        }
      } else {
        stageB(kn, Bsn, false);
      }
    }
  }

  // epilogue: C/D layout — VGPR r: lanes 0-15 -> M=r, lanes 16-31 -> M=r+8
#pragma unroll
  for (int s = 0; s < 2; ++s) {
#pragma unroll
    for (int t = 0; t < 4; ++t) {
      int col = n0 + t * 16 + l16;
      if (col < N) {
        float bv = bias ? bias[col] : 0.0f;
#pragma unroll
        for (int r = 0; r < 8; ++r) {
          int row = m0 + wave * 32 + s * 16 + hf * 8 + r;
          if (row < M) {
            float v = (s == 0 ? acc0[t][r] : acc1[t][r]) + bv;
            if (relu && v < 0.0f) v = 0.0f;
            C[(long)row * N + col] = v;
          }
        }
      }
    }
  }
}

// ---------------------------------------------------------------- batch norm
__global__ void bn_stats(const float* __restrict__ x, float* sum, float* ssq,
                         long n, int c) {
  long tot = n * c;
  long i = (long)blockIdx.x * blockDim.x + threadIdx.x;
  long st = (long)gridDim.x * blockDim.x;
  for (; i < tot; i += st) {
    float v = x[i];
    int ch = (int)(i % c);
    atomicAdd(&sum[ch], v);
    atomicAdd(&ssq[ch], v * v);
  }
}

__global__ void bn_apply(const float* __restrict__ x, const float* sum,
                         const float* ssq, const float* g, const float* b,
                         float* y, long n, int c) {
  long tot = n * c;
  long i = (long)blockIdx.x * blockDim.x + threadIdx.x;
  long st = (long)gridDim.x * blockDim.x;
  float inv_n = 1.0f / (float)n;
  for (; i < tot; i += st) {
    int ch = (int)(i % c);
    float mu = sum[ch] * inv_n;
    float var = ssq[ch] * inv_n - mu * mu;
    y[i] = (x[i] - mu) * rsqrtf(var + 1e-5f) * g[ch] + b[ch];
  }
}

// ---------------------------------------------------------------- attention
__global__ void edge_alpha(const float* __restrict__ Q, const float* __restrict__ Kn,
                           const float* __restrict__ We, const float* __restrict__ be,
                           const float* __restrict__ ea, const int* __restrict__ src,
                           const int* __restrict__ dst, float* alpha, float* amax,
                           int E) {
  int widx = (int)(((long)blockIdx.x * blockDim.x + threadIdx.x) >> 5);
  int lane = threadIdx.x & 31;
  int nw = (int)(((long)gridDim.x * blockDim.x) >> 5);
  for (int e = widx; e < E; e += nw) {
    int s = src[e], d = dst[e];
    float e0 = ea[e * 5 + 0], e1 = ea[e * 5 + 1], e2 = ea[e * 5 + 2];
    float e3 = ea[e * 5 + 3], e4 = ea[e * 5 + 4];
    float acc = 0.0f;
    for (int c = lane; c < HD; c += 32) {
      float ec = be[c] + We[c * 5 + 0] * e0 + We[c * 5 + 1] * e1 +
                 We[c * 5 + 2] * e2 + We[c * 5 + 3] * e3 + We[c * 5 + 4] * e4;
      acc += Q[(long)d * HD + c] * (Kn[(long)s * HD + c] + ec);
    }
    acc = warp_sum32(acc);
    if (lane == 0) {
      float a = acc * 0.04082482904638630f;  // 1/sqrt(600)
      alpha[e] = a;
      atomicMaxF(&amax[d], a);
    }
  }
}

__global__ void edge_exp(const float* alpha, const float* amax,
                         const int* __restrict__ dst, float* aexp, float* asum,
                         int E) {
  int i = blockIdx.x * blockDim.x + threadIdx.x;
  int st = gridDim.x * blockDim.x;
  for (; i < E; i += st) {
    float a = expf(alpha[i] - amax[dst[i]]);
    aexp[i] = a;
    atomicAdd(&asum[dst[i]], a);
  }
}

__global__ void edge_apply(const float* __restrict__ V, const float* __restrict__ We,
                           const float* __restrict__ be, const float* __restrict__ ea,
                           const float* aexp, const float* asum,
                           const int* __restrict__ src, const int* __restrict__ dst,
                           float* Hout, int E) {
  int widx = (int)(((long)blockIdx.x * blockDim.x + threadIdx.x) >> 5);
  int lane = threadIdx.x & 31;
  int nw = (int)(((long)gridDim.x * blockDim.x) >> 5);
  for (int e = widx; e < E; e += nw) {
    int s = src[e], d = dst[e];
    float w = aexp[e] / (asum[d] + 1e-16f);
    float e0 = ea[e * 5 + 0], e1 = ea[e * 5 + 1], e2 = ea[e * 5 + 2];
    float e3 = ea[e * 5 + 3], e4 = ea[e * 5 + 4];
    for (int c = lane; c < HD; c += 32) {
      float ec = be[c] + We[c * 5 + 0] * e0 + We[c * 5 + 1] * e1 +
                 We[c * 5 + 2] * e2 + We[c * 5 + 3] * e3 + We[c * 5 + 4] * e4;
      atomicAdd(&Hout[(long)d * HD + c], (V[(long)s * HD + c] + ec) * w);
    }
  }
}

// ---------------------------------------------------------------- GGC pieces
__global__ void scatter_add_feat(const float* __restrict__ msg,
                                 const int* __restrict__ src,
                                 const int* __restrict__ dst, float* agg, int E) {
  int widx = (int)(((long)blockIdx.x * blockDim.x + threadIdx.x) >> 5);
  int lane = threadIdx.x & 31;
  int nw = (int)(((long)gridDim.x * blockDim.x) >> 5);
  for (int e = widx; e < E; e += nw) {
    int s = src[e], d = dst[e];
    for (int c = lane; c < HD; c += 32)
      atomicAdd(&agg[(long)d * HD + c], msg[(long)s * HD + c]);
  }
}

__global__ void gru_gate(const float* __restrict__ gi, const float* __restrict__ gh,
                         const float* __restrict__ h, float* hout, int rows,
                         int base) {
  long tot = (long)rows * HD;
  long i = (long)blockIdx.x * blockDim.x + threadIdx.x;
  long st = (long)gridDim.x * blockDim.x;
  for (; i < tot; i += st) {
    int r = (int)(i / HD), j = (int)(i % HD);
    long gb = (long)r * 3 * HD;
    float ir = gi[gb + j], iz = gi[gb + HD + j], in = gi[gb + 2 * HD + j];
    float hr = gh[gb + j], hz = gh[gb + HD + j], hn = gh[gb + 2 * HD + j];
    float rr = 1.0f / (1.0f + expf(-(ir + hr)));
    float zz = 1.0f / (1.0f + expf(-(iz + hz)));
    float nn = tanhf(in + rr * hn);
    long hi = (long)(base + r) * HD + j;
    hout[hi] = (1.0f - zz) * nn + zz * h[hi];
  }
}

// ---------------------------------------------------------------- pooling etc
__global__ void aa_accum(const float* __restrict__ h, const int* __restrict__ aidx,
                         float* aasum, long tot) {
  long i = (long)blockIdx.x * blockDim.x + threadIdx.x;
  long st = (long)gridDim.x * blockDim.x;
  for (; i < tot; i += st) {
    int atom = (int)(i / HD), j = (int)(i % HD);
    atomicAdd(&aasum[(long)aidx[atom] * HD + j], h[i]);
  }
}

__global__ void aa_count(const int* __restrict__ aidx, float* cnt, int n) {
  int i = blockIdx.x * blockDim.x + threadIdx.x;
  int st = gridDim.x * blockDim.x;
  for (; i < n; i += st) atomicAdd(&cnt[aidx[i]], 1.0f);
}

__global__ void aa_div(float* aasum, const float* cnt, long tot) {
  long i = (long)blockIdx.x * blockDim.x + threadIdx.x;
  long st = (long)gridDim.x * blockDim.x;
  for (; i < tot; i += st) aasum[i] = aasum[i] / fmaxf(cnt[i / HD], 1.0f);
}

__global__ void concat_emb(const float* __restrict__ mean, const float* sum,
                           const float* ssq, const float* g, const float* b,
                           const float* __restrict__ emb, const int* __restrict__ pid,
                           float* cat, int n) {
  const int CW = HD + EMB;
  long tot = (long)n * CW;
  long i = (long)blockIdx.x * blockDim.x + threadIdx.x;
  long st = (long)gridDim.x * blockDim.x;
  float inv_n = 1.0f / (float)n;
  for (; i < tot; i += st) {
    int r = (int)(i / CW), j = (int)(i % CW);
    if (j < HD) {
      float mu = sum[j] * inv_n;
      float var = ssq[j] * inv_n - mu * mu;
      cat[i] = (mean[(long)r * HD + j] - mu) * rsqrtf(var + 1e-5f) * g[j] + b[j];
    } else {
      cat[i] = emb[(long)pid[r] * EMB + (j - HD)];
    }
  }
}

__global__ void pair_gather(const float* __restrict__ m, const int* __restrict__ eidx,
                            float* pair, int npair) {
  long tot = (long)npair * 2 * HD;
  long i = (long)blockIdx.x * blockDim.x + threadIdx.x;
  long st = (long)gridDim.x * blockDim.x;
  for (; i < tot; i += st) {
    int r = (int)(i / (2 * HD)), j = (int)(i % (2 * HD));
    int e = 2 * r;
    int node = (j < HD) ? eidx[e] : eidx[E_AA + e];
    int jj = (j < HD) ? j : j - HD;
    pair[i] = m[(long)node * HD + jj];
  }
}

__global__ void final_dot(const float* __restrict__ hid, const float* __restrict__ W2,
                          const float* __restrict__ b2, float* out, int n) {
  int widx = (int)(((long)blockIdx.x * blockDim.x + threadIdx.x) >> 5);
  int lane = threadIdx.x & 31;
  int nw = (int)(((long)gridDim.x * blockDim.x) >> 5);
  for (int i = widx; i < n; i += nw) {
    float acc = 0.0f;
    for (int c = lane; c < MLP_H; c += 32) acc += hid[(long)i * MLP_H + c] * W2[c];
    acc = warp_sum32(acc);
    if (lane == 0) out[i] = acc + b2[0];
  }
}

// ---------------------------------------------------------------- launcher
extern "C" void kernel_launch(void* const* d_in, const int* in_sizes, int n_in,
                              void* d_out, int out_size, void* d_ws, size_t ws_size,
                              hipStream_t stream) {
  const float* x   = (const float*)d_in[0];
  const float* ea  = (const float*)d_in[1];
  const int* iei   = (const int*)d_in[2];
  const int* aidx  = (const int*)d_in[3];
  const int* pid   = (const int*)d_in[4];
  const int* eidx  = (const int*)d_in[5];
  const float* bn1_g = (const float*)d_in[6];
  const float* bn1_b = (const float*)d_in[7];
  const float* Wq = (const float*)d_in[8];  const float* bq = (const float*)d_in[9];
  const float* Wk = (const float*)d_in[10]; const float* bk = (const float*)d_in[11];
  const float* Wv = (const float*)d_in[12]; const float* bv = (const float*)d_in[13];
  const float* We = (const float*)d_in[14]; const float* be = (const float*)d_in[15];
  const float* Wsk = (const float*)d_in[16]; const float* bsk = (const float*)d_in[17];
  const float* g1w = (const float*)d_in[18];
  const float* g1wih = (const float*)d_in[19]; const float* g1whh = (const float*)d_in[20];
  const float* g1bih = (const float*)d_in[21]; const float* g1bhh = (const float*)d_in[22];
  const float* bn2_g = (const float*)d_in[23]; const float* bn2_b = (const float*)d_in[24];
  const float* emb = (const float*)d_in[25];
  const float* Wp = (const float*)d_in[26]; const float* bp = (const float*)d_in[27];
  const float* g2w = (const float*)d_in[28];
  const float* g2wih = (const float*)d_in[29]; const float* g2whh = (const float*)d_in[30];
  const float* g2bih = (const float*)d_in[31]; const float* g2bhh = (const float*)d_in[32];
  const float* W1 = (const float*)d_in[33]; const float* b1 = (const float*)d_in[34];
  const float* W2 = (const float*)d_in[35]; const float* b2 = (const float*)d_in[36];
  float* out = (float*)d_out;

  const int* src = iei;
  const int* dst = iei + E_INNER;
  const int* s2 = eidx;
  const int* d2 = eidx + E_AA;

  // workspace carving (floats)
  float* ws = (float*)d_ws;
  long off = 0;
  auto alloc = [&](long n) { float* p = ws + off; off += (n + 63) & ~63L; return p; };
  float* SUM1 = alloc(64);  float* SSQ1 = alloc(64);
  float* SUM2 = alloc(640); float* SSQ2 = alloc(640);
  float* AMAX = alloc(N_ATOMS); float* ASUM = alloc(N_ATOMS);
  float* ALPHA = alloc(E_INNER); float* AEXP = alloc(E_INNER);
  float* CNT = alloc(N_AA);
  float* HBN = alloc((long)N_ATOMS * IN_C);
  float* B0 = alloc((long)N_ATOMS * HD);   // Q   -> agg
  float* B1 = alloc((long)N_ATOMS * HD);   // K   -> msg
  float* B2v = alloc((long)N_ATOMS * HD);  // V   -> h_new
  float* HB = alloc((long)N_ATOMS * HD);   // skip + attention out -> h
  const int CH = 12500;
  float* GI = alloc((long)CH * 3 * HD);
  float* GH = alloc((long)CH * 3 * HD);
  float* AAS = alloc((long)N_AA * HD);
  float* CAT = alloc((long)N_AA * (HD + EMB));
  float* MAA = alloc((long)N_AA * HD);
  float* MAA2 = alloc((long)N_AA * HD);
  float* AGG2 = alloc((long)N_AA * HD);
  float* MSG2 = alloc((long)N_AA * HD);
  const int NPAIR = E_AA / 2;
  float* PAIR = alloc((long)NPAIR * 2 * HD);
  float* HID = alloc((long)NPAIR * MLP_H);

  auto nb = [](long threads) {
    long b = (threads + 255) / 256;
    if (b < 1) b = 1;
    if (b > 16384) b = 16384;
    return (int)b;
  };
  auto gemm = [&](const float* A, const float* Bm, const float* bias, float* C,
                  int M, int N, int K, int ldb, int bT, int relu) {
    dim3 g((N + BN - 1) / BN, (M + BM - 1) / BM);
    gemm_wmma_f16<<<g, 256, 0, stream>>>(A, Bm, bias, C, M, N, K, ldb, bT, relu);
  };

  // ---- BatchNorm1d(42)
  fill_f32<<<nb(1408), 256, 0, stream>>>(SUM1, 0.0f, 64 + 64 + 640 + 640);
  bn_stats<<<nb((long)N_ATOMS * IN_C), 256, 0, stream>>>(x, SUM1, SSQ1, N_ATOMS, IN_C);
  bn_apply<<<nb((long)N_ATOMS * IN_C), 256, 0, stream>>>(x, SUM1, SSQ1, bn1_g, bn1_b,
                                                         HBN, N_ATOMS, IN_C);
  // ---- TransformerConv projections (x @ W.T + b)
  gemm(HBN, Wq, bq, B0, N_ATOMS, HD, IN_C, IN_C, 1, 0);
  gemm(HBN, Wk, bk, B1, N_ATOMS, HD, IN_C, IN_C, 1, 0);
  gemm(HBN, Wv, bv, B2v, N_ATOMS, HD, IN_C, IN_C, 1, 0);
  gemm(HBN, Wsk, bsk, HB, N_ATOMS, HD, IN_C, IN_C, 1, 0);  // skip term into H
  // ---- edge softmax attention, scatter into HB
  fill_f32<<<nb(N_ATOMS), 256, 0, stream>>>(AMAX, -3.0e38f, N_ATOMS);
  fill_f32<<<nb(N_ATOMS), 256, 0, stream>>>(ASUM, 0.0f, N_ATOMS);
  edge_alpha<<<nb((long)E_INNER * 32), 256, 0, stream>>>(B0, B1, We, be, ea, src, dst,
                                                         ALPHA, AMAX, E_INNER);
  edge_exp<<<nb(E_INNER), 256, 0, stream>>>(ALPHA, AMAX, dst, AEXP, ASUM, E_INNER);
  edge_apply<<<nb((long)E_INNER * 32), 256, 0, stream>>>(B2v, We, be, ea, AEXP, ASUM,
                                                         src, dst, HB, E_INNER);

  // ---- GatedGraphConv #1 (atom graph), 4 layers, row-chunked GRU GEMMs
  float* hcur = HB;
  float* hnew = B2v;
  for (int l = 0; l < NLAYERS; ++l) {
    fill_f32<<<nb((long)N_ATOMS * HD), 256, 0, stream>>>(B0, 0.0f, (long)N_ATOMS * HD);
    gemm(hcur, g1w + (long)l * HD * HD, nullptr, B1, N_ATOMS, HD, HD, HD, 0, 0);
    scatter_add_feat<<<nb((long)E_INNER * 32), 256, 0, stream>>>(B1, src, dst, B0, E_INNER);
    for (int r0 = 0; r0 < N_ATOMS; r0 += CH) {
      int rows = (N_ATOMS - r0 < CH) ? (N_ATOMS - r0) : CH;
      gemm(B0 + (long)r0 * HD, g1wih, g1bih, GI, rows, 3 * HD, HD, HD, 1, 0);
      gemm(hcur + (long)r0 * HD, g1whh, g1bhh, GH, rows, 3 * HD, HD, HD, 1, 0);
      gru_gate<<<nb((long)rows * HD), 256, 0, stream>>>(GI, GH, hcur, hnew, rows, r0);
    }
    float* t = hcur; hcur = hnew; hnew = t;
  }

  // ---- atoms -> amino acids mean pool
  fill_f32<<<nb((long)N_AA * HD), 256, 0, stream>>>(AAS, 0.0f, (long)N_AA * HD);
  fill_f32<<<nb(N_AA), 256, 0, stream>>>(CNT, 0.0f, N_AA);
  aa_accum<<<nb((long)N_ATOMS * HD), 256, 0, stream>>>(hcur, aidx, AAS, (long)N_ATOMS * HD);
  aa_count<<<nb(N_ATOMS), 256, 0, stream>>>(aidx, CNT, N_ATOMS);
  aa_div<<<nb((long)N_AA * HD), 256, 0, stream>>>(AAS, CNT, (long)N_AA * HD);
  // ---- bn2 + embedding concat + projection (relu)
  bn_stats<<<nb((long)N_AA * HD), 256, 0, stream>>>(AAS, SUM2, SSQ2, N_AA, HD);
  concat_emb<<<nb((long)N_AA * (HD + EMB)), 256, 0, stream>>>(AAS, SUM2, SSQ2, bn2_g,
                                                              bn2_b, emb, pid, CAT, N_AA);
  gemm(CAT, Wp, bp, MAA, N_AA, HD, HD + EMB, HD + EMB, 1, 1);

  // ---- GatedGraphConv #2 (amino-acid graph), 4 layers
  float* mcur = MAA;
  float* mnew = MAA2;
  for (int l = 0; l < NLAYERS; ++l) {
    fill_f32<<<nb((long)N_AA * HD), 256, 0, stream>>>(AGG2, 0.0f, (long)N_AA * HD);
    gemm(mcur, g2w + (long)l * HD * HD, nullptr, MSG2, N_AA, HD, HD, HD, 0, 0);
    scatter_add_feat<<<nb((long)E_AA * 32), 256, 0, stream>>>(MSG2, s2, d2, AGG2, E_AA);
    gemm(AGG2, g2wih, g2bih, GI, N_AA, 3 * HD, HD, HD, 1, 0);
    gemm(mcur, g2whh, g2bhh, GH, N_AA, 3 * HD, HD, HD, 1, 0);
    gru_gate<<<nb((long)N_AA * HD), 256, 0, stream>>>(GI, GH, mcur, mnew, N_AA, 0);
    float* t = mcur; mcur = mnew; mnew = t;
  }

  // ---- MLP head on every-other edge pair
  pair_gather<<<nb((long)NPAIR * 2 * HD), 256, 0, stream>>>(mcur, eidx, PAIR, NPAIR);
  gemm(PAIR, W1, b1, HID, NPAIR, MLP_H, 2 * HD, 2 * HD, 1, 1);
  final_dot<<<nb((long)NPAIR * 32), 256, 0, stream>>>(HID, W2, b2, out, NPAIR);
}
